// WaveNet_VNNs_49598282334409
// MI455X (gfx1250) — matrix operand order repaired and checked by reading, not compiled, not run on hardware
//
#include <hip/hip_runtime.h>
#include <math.h>

// ---------------------------------------------------------------------------
// WaveNet + VNN forward for MI455X (gfx1250, wave32, WMMA).
// Dilated causal convs -> GEMM on v_wmma_f32_16x16x32_f16:
//   G[128 x Ttile] = W[128 x 192] * Hext[192 x Ttile],
// Hext row r = k*64+ci holds h[ci, t - d*(2-k)].
// - Hext image in LDS transposed & K-contiguous (XT[col][r], stride 200 halfs)
//   => each lane's B fragment is two ds_load_b128, zero packing VALU.
// - Pre-activations in LDS transposed (Gt[col][M], stride 136 halfs)
//   => C-tile writeback is 4x v_cvt_pk_f16_f32 + one ds_store_b128.
// - Weights pre-converted once to f16 in WMMA A-fragment order
//   => A fragment = two global_load_b128, no runtime conversion.
// ---------------------------------------------------------------------------

typedef __attribute__((ext_vector_type(16))) _Float16 v16h;
typedef __attribute__((ext_vector_type(8)))  _Float16 v8h;
typedef __attribute__((ext_vector_type(2)))  _Float16 v2h;
typedef __attribute__((ext_vector_type(8)))  float    v8f;

#define T_LEN 32000
#define BATCH 8
#define RESC  64
#define SKIPC 64
#define TT    128      // output timesteps per workgroup
#define CB    144      // tile columns incl. 16-col left halo (skip-conv taps)
#define KR    192      // reduction = 64 channels * 3 taps
#define SR    200      // XT row stride (halfs): 400B, 16B-aligned, 100-dword
                       // stride => conflict-free b128 fragment loads
#define GSR   136      // Gt row stride (halfs): 272B, 16B-aligned, 68-dword
                       // stride => conflict-free b128 C-tile stores
#define FRAG  3072     // halfs per m-tile of fragment-ordered weights (6*32*16)

__device__ __forceinline__ float sigmoidf_(float x) {
  return 1.0f / (1.0f + __expf(-x));
}

__device__ __forceinline__ v16h cat8(v8h lo, v8h hi) {
  return __builtin_shufflevector(lo, hi, 0, 1, 2, 3, 4, 5, 6, 7,
                                 8, 9, 10, 11, 12, 13, 14, 15);
}

// ---- weight prep: f32 [co][ci][k] -> f16 WMMA A-fragment order -------------
// dst layout: [mt][kc][lane][e(16)]; element e of lane's v16h (contiguous k, same order as load_B_xt):
// M = mt*16 + (lane&15), hi = lane>>4, j = e>>1,
// kofs = 2j+16hi, r = kc*32 + kofs + (e&1),
// src = W[M*192 + (r%64)*3 + r/64].
__global__ void prep_w_kernel(const float* __restrict__ W,
                              _Float16* __restrict__ dst, int n)
{
  const int idx = blockIdx.x * 256 + threadIdx.x;
  if (idx >= n) return;
  const int e    = idx & 15;
  const int lane = (idx >> 4) & 31;
  const int kc   = (idx >> 9) % 6;
  const int mt   = idx / FRAG;
  const int M    = mt * 16 + (lane & 15);
  const int hi   = lane >> 4;
  const int j    = e >> 1;
  const int kofs = 2 * j + 16 * hi;
  const int r    = kc * 32 + kofs + (e & 1);
  dst[idx] = (_Float16)W[M * KR + (r & 63) * 3 + (r >> 6)];
}

// A fragment from fragment-ordered f16 weights: two global_load_b128.
__device__ __forceinline__ v16h load_A_f(const _Float16* __restrict__ WF,
                                         int mt, int kc) {
  const int lane = threadIdx.x & 31;
  const _Float16* p = WF + ((mt * 6 + kc) * 32 + lane) * 16;  // 32B/lane
  return cat8(*(const v8h*)p, *(const v8h*)(p + 8));
}

// B-matrix 32x16 f16 fragment from transposed LDS image XT[col][r]:
// lane's 16 values are K-contiguous -> two ds_load_b128.
__device__ __forceinline__ v16h load_B_xt(const _Float16* XT, int ncol, int kc) {
  const int lane = threadIdx.x & 31;
  const int n  = ncol + (lane & 15);
  const int kb = kc * 32 + (lane >> 4) * 16;
  const _Float16* p = XT + n * SR + kb;   // 16B-aligned (SR*2=400, kb*2%32==0)
  return cat8(*(const v8h*)p, *(const v8h*)(p + 8));
}

// ---- gated residual layer --------------------------------------------------
__global__ __launch_bounds__(256) void layer_kernel(
    const float* __restrict__ h_in, float* __restrict__ h_out,
    float* __restrict__ skip,
    const _Float16* __restrict__ wresf, const _Float16* __restrict__ wskipf,
    int dil)
{
  // XT: first the dilation-shifted input image, then (after matmul 1) reused
  // for the k-shifted gated-activation image feeding the skip conv.
  __shared__ __align__(16) _Float16 XT[CB * SR];    // 57600 B
  __shared__ __align__(16) _Float16 Gt[CB * GSR];   // 39168 B pre-activations
  // total 96768 B -> 3 workgroups per 320KB WGP

  const int tid  = threadIdx.x;
  const int wave = tid >> 5;
  const int lane = tid & 31;
  const int b    = blockIdx.y;
  const int t0   = blockIdx.x * TT;

  // Stage dilation-shifted input tile (coalesced global f32 -> transposed LDS f16).
  for (int idx = tid; idx < KR * CB; idx += 256) {
    const int r = idx / CB, col = idx % CB;
    const int k = r >> 6, ci = r & 63;
    const int t = t0 - 16 + col - dil * (2 - k);
    XT[col * SR + r] =
        (_Float16)((t >= 0) ? h_in[(b * RESC + ci) * T_LEN + t] : 0.0f);
  }
  __syncthreads();

  // ---- matmul 1: Gt = (W_res[128x192] * Hext[192x144])^T; wave -> m-tile.
  {
    v16h aw[6];
#pragma unroll
    for (int kc = 0; kc < 6; ++kc) aw[kc] = load_A_f(wresf, wave, kc);
    for (int nt = 0; nt < 9; ++nt) {
      v8f acc = {};
#pragma unroll
      for (int kc = 0; kc < 6; ++kc) {
        v16h bx = load_B_xt(XT, nt * 16, kc);
        acc = __builtin_amdgcn_wmma_f32_16x16x32_f16(
            false, aw[kc], false, bx, (short)0, acc, false, false);
      }
      const int m0 = wave * 16 + 8 * (lane >> 4);
      const int n  = nt * 16 + (lane & 15);
      v8h ch;
#pragma unroll
      for (int r = 0; r < 8; ++r) ch[r] = (_Float16)acc[r];
      *(v8h*)&Gt[n * GSR + m0] = ch;    // one ds_store_b128
    }
  }
  __syncthreads();

  // gated = tanh(Gt[:, :64]) * sigmoid(Gt[:, 64:]); scatter ci-pairs into the
  // 3 k-shifted positions of the transposed image (XT reused, Hext dead).
  for (int idx = tid; idx < (RESC / 2) * CB; idx += 256) {
    const int cp  = idx & 31;           // channel pair: ci = 2*cp, 2*cp+1
    const int col = idx >> 5;
    const v2h ta = *(const v2h*)&Gt[col * GSR + 2 * cp];
    const v2h sa = *(const v2h*)&Gt[col * GSR + RESC + 2 * cp];
    v2h g;
    g[0] = (_Float16)(tanhf((float)ta[0]) * sigmoidf_((float)sa[0]));
    g[1] = (_Float16)(tanhf((float)ta[1]) * sigmoidf_((float)sa[1]));
#pragma unroll
    for (int k = 0; k < 3; ++k) {
      const int j = col + (2 - k);      // output column this tap feeds
      if (j < CB) *(v2h*)&XT[j * SR + (k * 64 + 2 * cp)] = g;
    }
  }
  __syncthreads();

  // ---- matmul 2: O = W_skip[128x192] * GatedExt; residual + skip update.
  {
    v16h aw[6];
#pragma unroll
    for (int kc = 0; kc < 6; ++kc) aw[kc] = load_A_f(wskipf, wave, kc);
    for (int nt = 0; nt < 8; ++nt) {
      v8f acc = {};
#pragma unroll
      for (int kc = 0; kc < 6; ++kc) {
        v16h bx = load_B_xt(XT, 16 + nt * 16, kc);
        acc = __builtin_amdgcn_wmma_f32_16x16x32_f16(
            false, aw[kc], false, bx, (short)0, acc, false, false);
      }
      const int co0 = wave * 16 + 8 * (lane >> 4);
      const int t   = t0 + nt * 16 + (lane & 15);
#pragma unroll
      for (int r = 0; r < 8; ++r) {
        const int co = co0 + r;
        if (co < RESC) {
          h_out[(b * RESC + co) * T_LEN + t] =
              h_in[(b * RESC + co) * T_LEN + t] + acc[r];
        } else {
          float* sp = &skip[(b * SKIPC + (co - RESC)) * T_LEN + t];
          *sp = *sp + acc[r];
        }
      }
    }
  }
}

// ---- 64->64 k=3 causal conv with tanh output (and optional input tanh) -----
__global__ __launch_bounds__(256) void conv64_kernel(
    const float* __restrict__ in, float* __restrict__ out,
    const _Float16* __restrict__ wf, int in_tanh)
{
  __shared__ __align__(16) _Float16 XT[CB * SR];   // 57600 B

  const int tid  = threadIdx.x;
  const int wave = tid >> 5;
  const int lane = tid & 31;
  const int b    = blockIdx.y;
  const int t0   = blockIdx.x * TT;

  // Build the k-shifted transposed image directly at staging (dilation 1).
  for (int idx = tid; idx < KR * CB; idx += 256) {
    const int r = idx / CB, col = idx % CB;
    const int k = r >> 6, ci = r & 63;
    const int t = t0 - 16 + col - (2 - k);
    float v = (t >= 0 && t < T_LEN) ? in[(b * RESC + ci) * T_LEN + t] : 0.0f;
    if (in_tanh) v = tanhf(v);
    XT[col * SR + r] = (_Float16)v;
  }
  __syncthreads();

  // 4 m-tiles x 8 n-tiles over 8 waves: wave&3 -> m-tile, wave>>2 -> n half.
  const int mt = wave & 3;
  const int nh = wave >> 2;
  v16h aw[6];
#pragma unroll
  for (int kc = 0; kc < 6; ++kc) aw[kc] = load_A_f(wf, mt, kc);
  for (int nt = nh * 4; nt < nh * 4 + 4; ++nt) {
    v8f acc = {};
#pragma unroll
    for (int kc = 0; kc < 6; ++kc) {
      v16h bx = load_B_xt(XT, 16 + nt * 16, kc);
      acc = __builtin_amdgcn_wmma_f32_16x16x32_f16(
          false, aw[kc], false, bx, (short)0, acc, false, false);
    }
    const int co0 = mt * 16 + 8 * (lane >> 4);
    const int t   = t0 + nt * 16 + (lane & 15);
#pragma unroll
    for (int r = 0; r < 8; ++r)
      out[(b * RESC + co0 + r) * T_LEN + t] = tanhf(acc[r]);
  }
}

// ---- front conv: x[8,1,T] -> h[8,64,T] -------------------------------------
__global__ void conv_in_kernel(const float* __restrict__ x,
                               const float* __restrict__ w_in,
                               float* __restrict__ h)
{
  const int idx = blockIdx.x * 256 + threadIdx.x;
  if (idx >= BATCH * RESC * T_LEN) return;
  const int t  = idx % T_LEN;
  const int co = (idx / T_LEN) % RESC;
  const int b  = idx / (T_LEN * RESC);
  float acc = 0.0f;
#pragma unroll
  for (int j = 0; j < 3; ++j) {
    const int ts = t - (2 - j);
    if (ts >= 0) acc += w_in[co * 3 + j] * x[b * T_LEN + ts];
  }
  h[idx] = acc;
}

// ---- head: y4 = tanh(cconv(y3, w4)) ----------------------------------------
__global__ void head1_kernel(const float* __restrict__ y3,
                             const float* __restrict__ w4,
                             float* __restrict__ y4)
{
  const int idx = blockIdx.x * 256 + threadIdx.x;
  if (idx >= BATCH * T_LEN) return;
  const int t = idx % T_LEN, b = idx / T_LEN;
  float acc = 0.0f;
  for (int ci = 0; ci < RESC; ++ci) {
    const float* row = &y3[(b * RESC + ci) * T_LEN];
#pragma unroll
    for (int j = 0; j < 3; ++j) {
      const int ts = t - (2 - j);
      if (ts >= 0) acc += w4[ci * 3 + j] * row[ts];
    }
  }
  y4[idx] = tanhf(acc);
}

// ---- VNN2: lin + 2nd-order Volterra ----------------------------------------
__global__ void head2_kernel(const float* __restrict__ y4,
                             const float* __restrict__ vw1,
                             const float* __restrict__ vw2,
                             float* __restrict__ out)
{
  const int idx = blockIdx.x * 256 + threadIdx.x;
  if (idx >= BATCH * T_LEN) return;
  const int t = idx % T_LEN, b = idx / T_LEN;
  float yv[5];
#pragma unroll
  for (int j = 0; j < 5; ++j) {
    const int ts = t - (4 - j);
    yv[j] = (ts >= 0) ? y4[b * T_LEN + ts] : 0.0f;
  }
  float lin = 0.0f;
#pragma unroll
  for (int j = 0; j < 5; ++j) lin += vw1[j] * yv[j];
  float x2[6];
#pragma unroll
  for (int c = 0; c < 6; ++c) {
    float s = 0.0f;
#pragma unroll
    for (int j = 0; j < 5; ++j) s += vw2[c * 5 + j] * yv[j];
    x2[c] = s;
  }
  const float quad = x2[0] * x2[3] + x2[1] * x2[4] + x2[2] * x2[5];
  out[idx] = lin + quad;
}

// ---------------------------------------------------------------------------
extern "C" void kernel_launch(void* const* d_in, const int* in_sizes, int n_in,
                              void* d_out, int out_size, void* d_ws, size_t ws_size,
                              hipStream_t stream)
{
  (void)in_sizes; (void)n_in; (void)out_size; (void)ws_size;

  const float* x      = (const float*)d_in[0];
  const float* w_in   = (const float*)d_in[1];
  const float* w_res  = (const float*)d_in[2];
  const float* w_skip = (const float*)d_in[3];
  const float* w2     = (const float*)d_in[4];
  const float* w3     = (const float*)d_in[5];
  const float* w4     = (const float*)d_in[6];
  const float* vw1    = (const float*)d_in[7];
  const float* vw2    = (const float*)d_in[8];
  float* out = (float*)d_out;

  const size_t plane = (size_t)BATCH * RESC * T_LEN;   // 16.384 M floats
  float* hA   = (float*)d_ws;
  float* hB   = hA + plane;
  float* skip = hB + plane;
  float* y4   = skip + plane;                          // BATCH*T_LEN floats
  // f16 fragment-ordered weights (one-time prep, L2-resident ~1MB)
  _Float16* wresf  = (_Float16*)(y4 + (size_t)BATCH * T_LEN);
  _Float16* wskipf = wresf  + (size_t)10 * 8 * FRAG;
  _Float16* w2f    = wskipf + (size_t)10 * 8 * FRAG;
  _Float16* w3f    = w2f + 4 * FRAG;

  hipMemsetAsync(skip, 0, plane * sizeof(float), stream);

  // one-time weight conversion to WMMA fragment order
  for (int i = 0; i < 10; ++i) {
    const int n = 8 * FRAG;
    prep_w_kernel<<<(n + 255) / 256, 256, 0, stream>>>(
        w_res + (size_t)i * 128 * 64 * 3, wresf + (size_t)i * 8 * FRAG, n);
    prep_w_kernel<<<(n + 255) / 256, 256, 0, stream>>>(
        w_skip + (size_t)i * 128 * 64 * 3, wskipf + (size_t)i * 8 * FRAG, n);
  }
  {
    const int n = 4 * FRAG;
    prep_w_kernel<<<(n + 255) / 256, 256, 0, stream>>>(w2, w2f, n);
    prep_w_kernel<<<(n + 255) / 256, 256, 0, stream>>>(w3, w3f, n);
  }

  {
    const int n = BATCH * RESC * T_LEN;
    conv_in_kernel<<<(n + 255) / 256, 256, 0, stream>>>(x, w_in, hA);
  }

  dim3 grid(T_LEN / TT, BATCH);
  float* cur = hA;
  float* nxt = hB;
  for (int s = 0; s < 2; ++s) {
    for (int i = 0; i < 10; ++i) {
      layer_kernel<<<grid, 256, 0, stream>>>(
          cur, nxt, skip,
          wresf  + (size_t)i * 8 * FRAG,
          wskipf + (size_t)i * 8 * FRAG,
          1 << i);
      float* tmp = cur; cur = nxt; nxt = tmp;
    }
  }
  // 20 swaps -> cur == hA (final h, now dead), hB free.
  conv64_kernel<<<grid, 256, 0, stream>>>(skip, hB, w2f, 1);  // tanh(skip)->w2->tanh
  conv64_kernel<<<grid, 256, 0, stream>>>(hB, hA, w3f, 0);    // w3 -> tanh

  {
    const int n = BATCH * T_LEN;
    head1_kernel<<<(n + 255) / 256, 256, 0, stream>>>(hA, w4, y4);
    head2_kernel<<<(n + 255) / 256, 256, 0, stream>>>(y4, vw1, vw2, out);
  }
}
